// GCSAN_46351287058915
// MI455X (gfx1250) — compile-verified, matrix operand order checked by reading, not run on hardware
//
#include <hip/hip_runtime.h>
#include <hip/hip_bf16.h>

typedef __attribute__((ext_vector_type(16))) __bf16 v16bf;
typedef __attribute__((ext_vector_type(8)))  float  v8f;

#define NV_ITEMS 100000
#define SL  50
#define DIM 128
#define BATCH 512
#define NHD 2
#define HDD 64
#define BL  (BATCH * SL)        /* 25600 */
#define BLD (BL * DIM)          /* 3276800 */

// Pack 16 f32 (two contiguous 8-float runs) into a bf16 fragment.
__device__ __forceinline__ void pack16(v16bf& d, const float4& q0, const float4& q1,
                                       const float4& q2, const float4& q3)
{
    d[0]  = (__bf16)q0.x; d[1]  = (__bf16)q0.y; d[2]  = (__bf16)q0.z; d[3]  = (__bf16)q0.w;
    d[4]  = (__bf16)q1.x; d[5]  = (__bf16)q1.y; d[6]  = (__bf16)q1.z; d[7]  = (__bf16)q1.w;
    d[8]  = (__bf16)q2.x; d[9]  = (__bf16)q2.y; d[10] = (__bf16)q2.z; d[11] = (__bf16)q2.w;
    d[12] = (__bf16)q3.x; d[13] = (__bf16)q3.y; d[14] = (__bf16)q3.z; d[15] = (__bf16)q3.w;
}

// ---------------------------------------------------------------------------
// WMMA GEMM, compile-time K and B-layout:  C[M,N] = act( A[M,K] @ B + bias + addC )
//   BTRANS=false: B[k*N + n]  (row-major K x N weight)
//   BTRANS=true : B[n*K + k]  (reps @ item_emb^T)
// One wave per 16x16 C tile, fully unrolled K loop (K/32 back-to-back WMMAs).
// ISA bf16 A/B fragment layout: element i of lane -> k = (i/8)*16 + (lane/16)*8 + i%8,
// i.e. two contiguous 8-float runs per lane -> 4x b128 loads per 32-K step.
// ---------------------------------------------------------------------------
template <int KK, bool BTRANS>
__global__ __launch_bounds__(256)
void gemm_wmma(const float* __restrict__ A, const float* __restrict__ Bm,
               const float* __restrict__ bias, const float* __restrict__ addC,
               float* __restrict__ C, int M, int N, int act)
{
    const int wave = threadIdx.x >> 5;
    const int lane = threadIdx.x & 31;
    const int Mt = M >> 4;
    const int Nt = (N + 15) >> 4;
    const int tile = blockIdx.x * 8 + wave;
    if (tile >= Mt * Nt) return;                   // wave-uniform exit
    const int tm = (tile % Mt) << 4;
    const int tn = (tile / Mt) << 4;
    const int hf  = lane >> 4;                     // lane half: 0 / 1
    const int l16 = lane & 15;
    const int arow = tm + l16;
    const int bcol = tn + l16;
    const int bcolc = bcol < N ? bcol : N - 1;     // clamp loads at N edge

    const float* aprow = A + (size_t)arow * KK + (hf << 3);
    const float* bprow = BTRANS ? (Bm + (size_t)bcolc * KK + (hf << 3)) : Bm;

    v8f acc = {};
#pragma unroll
    for (int kk = 0; kk < KK; kk += 32) {
        v16bf af, bfv;
        {   // A fragment: two contiguous 8-float runs -> 4x float4
            const float* ap = aprow + kk;
            const float4 a0 = *(const float4*)(ap);
            const float4 a1 = *(const float4*)(ap + 4);
            const float4 a2 = *(const float4*)(ap + 16);
            const float4 a3 = *(const float4*)(ap + 20);
            pack16(af, a0, a1, a2, a3);
        }
        if (BTRANS) {   // B row-contiguous in k: same vectorized pattern
            const float* bp = bprow + kk;
            const float4 b0 = *(const float4*)(bp);
            const float4 b1 = *(const float4*)(bp + 4);
            const float4 b2 = *(const float4*)(bp + 16);
            const float4 b3 = *(const float4*)(bp + 20);
            pack16(bfv, b0, b1, b2, b3);
        } else {        // weight matrix: strided by N (L2-resident, coalesced over lanes)
#pragma unroll
            for (int i = 0; i < 16; ++i) {
                const int kl = ((i >> 3) << 4) + (hf << 3) + (i & 7);
                bfv[i] = (__bf16)Bm[(size_t)(kk + kl) * N + bcolc];
            }
        }
        acc = __builtin_amdgcn_wmma_f32_16x16x32_bf16(
                  false, af, false, bfv, (short)0, acc, false, false);
    }

    if (bcol < N) {
        const float bb = bias ? bias[bcol] : 0.0f;
#pragma unroll
        for (int v = 0; v < 8; ++v) {
            const int row = tm + (hf << 3) + v;
            float r = acc[v] + bb;
            if (addC) r += addC[(size_t)row * N + bcol];
            if      (act == 1) r = r > 0.0f ? r : 0.0f;
            else if (act == 2) r = 1.0f / (1.0f + __expf(-r));
            else if (act == 3) r = tanhf(r);
            C[(size_t)row * N + bcol] = r;
        }
    }
}

// ---------------------------------------------------------------------------
// Embedding gather + positional add + LayerNorm(ln0). One block per token.
// ---------------------------------------------------------------------------
__global__ __launch_bounds__(128)
void embed_ln0(const int* __restrict__ seq, const float* __restrict__ item_emb,
               const float* __restrict__ pos_emb, const float* __restrict__ g,
               const float* __restrict__ b, float* __restrict__ X)
{
    __shared__ float red[DIM];
    const int t = blockIdx.x;
    const int l = t % SL;
    const int d = threadIdx.x;
    const int it = seq[t];
    float x = item_emb[(size_t)it * DIM + d] + pos_emb[l * DIM + d];
    red[d] = x; __syncthreads();
    for (int s = 64; s > 0; s >>= 1) { if (d < s) red[d] += red[d + s]; __syncthreads(); }
    const float mean = red[0] / DIM; __syncthreads();
    const float c = x - mean;
    red[d] = c * c; __syncthreads();
    for (int s = 64; s > 0; s >>= 1) { if (d < s) red[d] += red[d + s]; __syncthreads(); }
    const float var = red[0] / DIM;
    X[(size_t)t * DIM + d] = c * rsqrtf(var + 1e-5f) * g[d] + b[d];
}

// LayerNorm of (a + r) per row (r may be null). In-place safe (out may == a).
__global__ __launch_bounds__(128)
void ln_res(const float* __restrict__ a, const float* __restrict__ r,
            const float* __restrict__ g, const float* __restrict__ b,
            float* __restrict__ out)
{
    __shared__ float red[DIM];
    const int t = blockIdx.x;
    const int d = threadIdx.x;
    const size_t idx = (size_t)t * DIM + d;
    float x = a[idx] + (r ? r[idx] : 0.0f);
    red[d] = x; __syncthreads();
    for (int s = 64; s > 0; s >>= 1) { if (d < s) red[d] += red[d + s]; __syncthreads(); }
    const float mean = red[0] / DIM; __syncthreads();
    const float c = x - mean;
    red[d] = c * c; __syncthreads();
    for (int s = 64; s > 0; s >>= 1) { if (d < s) red[d] += red[d + s]; __syncthreads(); }
    const float var = red[0] / DIM;
    out[idx] = c * rsqrtf(var + 1e-5f) * g[d] + b[d];
}

// ---------------------------------------------------------------------------
// 2-head attention over L=50. One 64-thread block per (b,h).
// ---------------------------------------------------------------------------
__global__ __launch_bounds__(64)
void attn_kernel(const float* __restrict__ Q, const float* __restrict__ Kt,
                 const float* __restrict__ Vt, float* __restrict__ AO)
{
    __shared__ float qb[SL * HDD], kb[SL * HDD], vb[SL * HDD];
    const int b = blockIdx.x / NHD;
    const int h = blockIdx.x % NHD;
    const size_t base = (size_t)b * SL * DIM + h * HDD;
    for (int i = threadIdx.x; i < SL * HDD; i += 64) {
        const int m = i / HDD, d = i % HDD;
        const size_t src = base + (size_t)m * DIM + d;
        qb[i] = Q[src]; kb[i] = Kt[src]; vb[i] = Vt[src];
    }
    __syncthreads();
    const int l = threadIdx.x;
    if (l < SL) {
        float sc[SL];
        float mx = -1e30f;
        for (int m = 0; m < SL; ++m) {
            float s = 0.0f;
            for (int d = 0; d < HDD; ++d) s += qb[l * HDD + d] * kb[m * HDD + d];
            s *= 0.125f;                               // 1/sqrt(64)
            sc[m] = s; mx = fmaxf(mx, s);
        }
        float den = 0.0f;
        for (int m = 0; m < SL; ++m) { sc[m] = __expf(sc[m] - mx); den += sc[m]; }
        const float rden = 1.0f / den;
        for (int d = 0; d < HDD; ++d) {
            float o = 0.0f;
            for (int m = 0; m < SL; ++m) o += sc[m] * vb[m * HDD + d];
            AO[base + (size_t)l * DIM + d] = o * rden;
        }
    }
}

// ---------------------------------------------------------------------------
// GGNN graph build: stable sort, rank/inv/uniq, normalized in/out adjacency.
// ---------------------------------------------------------------------------
__global__ __launch_bounds__(64)
void ggnn_build(const int* __restrict__ seq, int* __restrict__ uniq,
                int* __restrict__ invb, float* __restrict__ A_in,
                float* __restrict__ A_out)
{
    const int s = blockIdx.x;
    __shared__ int sh_inv[SL];
    __shared__ int sh_uniq[SL];
    float* Ain  = A_in  + (size_t)s * SL * SL;
    float* Aout = A_out + (size_t)s * SL * SL;
    for (int i = threadIdx.x; i < SL * SL; i += 64) { Ain[i] = 0.0f; Aout[i] = 0.0f; }
    if (threadIdx.x == 0) {
        int val[SL], ord[SL];
        for (int i = 0; i < SL; ++i) { val[i] = seq[s * SL + i]; ord[i] = i; }
        for (int i = 1; i < SL; ++i) {          // stable insertion sort
            const int ov = ord[i]; const int vv = val[ov];
            int j = i - 1;
            while (j >= 0 && val[ord[j]] > vv) { ord[j + 1] = ord[j]; --j; }
            ord[j + 1] = ov;
        }
        for (int i = 0; i < SL; ++i) sh_uniq[i] = 0;   // item 0 -> zero emb row
        int rank = -1, prev = -1;
        for (int i = 0; i < SL; ++i) {
            const int vv = val[ord[i]];
            if (i == 0 || vv != prev) { ++rank; sh_uniq[rank] = vv; prev = vv; }
            sh_inv[ord[i]] = rank;
        }
    }
    __syncthreads();
    if (threadIdx.x == 0) {
        for (int i = 0; i < SL - 1; ++i) {
            const int src = sh_inv[i], dst = sh_inv[i + 1];
            Ain[dst * SL + src]  += 1.0f;
            Aout[src * SL + dst] += 1.0f;
        }
    }
    __syncthreads();
    for (int r = threadIdx.x; r < SL; r += 64) {
        float s1 = 0.0f, s2 = 0.0f;
        for (int c = 0; c < SL; ++c) { s1 += Ain[r * SL + c]; s2 += Aout[r * SL + c]; }
        const float i1 = 1.0f / (s1 + 1e-8f), i2 = 1.0f / (s2 + 1e-8f);
        for (int c = 0; c < SL; ++c) { Ain[r * SL + c] *= i1; Aout[r * SL + c] *= i2; }
        uniq[s * SL + r] = sh_uniq[r];
        invb[s * SL + r] = sh_inv[r];
    }
}

// H[r,:] = item_emb[uniq[r],:]
__global__ void gather_emb(const int* __restrict__ uniq,
                           const float* __restrict__ item_emb,
                           float* __restrict__ Hh, int n)
{
    const int i = blockIdx.x * blockDim.x + threadIdx.x;
    if (i >= n) return;
    const int r = i >> 7, d = i & 127;
    Hh[i] = item_emb[(size_t)uniq[r] * DIM + d];
}

// M = A_in @ hwin + A_out @ hwout  (per session, 50x50 @ 50x128). Block/(s,l).
__global__ __launch_bounds__(128)
void propagate(const float* __restrict__ A_in, const float* __restrict__ A_out,
               const float* __restrict__ hwin, const float* __restrict__ hwout,
               float* __restrict__ Mb)
{
    __shared__ float ai[SL], ao[SL];
    const int s = blockIdx.x / SL;
    const int l = blockIdx.x % SL;
    const float* Ain  = A_in  + ((size_t)s * SL + l) * SL;
    const float* Aout = A_out + ((size_t)s * SL + l) * SL;
    if (threadIdx.x < SL) { ai[threadIdx.x] = Ain[threadIdx.x]; ao[threadIdx.x] = Aout[threadIdx.x]; }
    __syncthreads();
    const int d = threadIdx.x;
    const float* hi = hwin  + (size_t)s * SL * DIM + d;
    const float* ho = hwout + (size_t)s * SL * DIM + d;
    float acc = 0.0f;
    for (int j = 0; j < SL; ++j) acc += ai[j] * hi[(size_t)j * DIM] + ao[j] * ho[(size_t)j * DIM];
    Mb[((size_t)s * SL + l) * DIM + d] = acc;
}

__global__ void ew_mul(const float* __restrict__ a, const float* __restrict__ b,
                       float* __restrict__ o, int n)
{ const int i = blockIdx.x * blockDim.x + threadIdx.x; if (i < n) o[i] = a[i] * b[i]; }

// h = (1-z)*h + z*ht
__global__ void ew_gru(const float* __restrict__ z, const float* __restrict__ ht,
                       float* __restrict__ h, int n)
{ const int i = blockIdx.x * blockDim.x + threadIdx.x;
  if (i < n) { const float zz = z[i]; h[i] = (1.0f - zz) * h[i] + zz * ht[i]; } }

// o = g*a + (1-g)*b
__global__ void ew_lerp(const float* __restrict__ g, const float* __restrict__ a,
                        const float* __restrict__ b, float* __restrict__ o, int n)
{ const int i = blockIdx.x * blockDim.x + threadIdx.x;
  if (i < n) { const float gg = g[i]; o[i] = gg * a[i] + (1.0f - gg) * b[i]; } }

// HL[s,l,:] = H[s, inv[s,l], :]
__global__ void gather_rows(const int* __restrict__ invb, const float* __restrict__ Hh,
                            float* __restrict__ HL, int n)
{
    const int i = blockIdx.x * blockDim.x + threadIdx.x;
    if (i >= n) return;
    const int r = i >> 7, d = i & 127;
    const int s = r / SL;
    HL[i] = Hh[((size_t)s * SL + invb[r]) * DIM + d];
}

// CAT[r, 0:128]=HL[r], CAT[r,128:256]=G[r]
__global__ void concat_cat(const float* __restrict__ HL, const float* __restrict__ G,
                           float* __restrict__ CAT, int n)
{
    const int i = blockIdx.x * blockDim.x + threadIdx.x;
    if (i >= n) return;
    const int r = i >> 8, d = i & 255;
    CAT[i] = (d < DIM) ? HL[(size_t)r * DIM + d] : G[(size_t)r * DIM + (d - DIM)];
}

// Session soft-attention readout -> [sg | last] (256 per session).
__global__ __launch_bounds__(128)
void sess_att(const float* __restrict__ HC, float* __restrict__ REPCAT)
{
    __shared__ float sc[SL];
    const int b = blockIdx.x;
    const float* hc   = HC + (size_t)b * SL * DIM;
    const float* last = hc + (size_t)(SL - 1) * DIM;
    const int t = threadIdx.x;
    if (t < SL) {
        float s = 0.0f;
        for (int d = 0; d < DIM; ++d) s += hc[(size_t)t * DIM + d] * last[d];
        sc[t] = s;
    }
    __syncthreads();
    if (t == 0) {
        float mx = sc[0];
        for (int i = 1; i < SL; ++i) mx = fmaxf(mx, sc[i]);
        float den = 0.0f;
        for (int i = 0; i < SL; ++i) { sc[i] = __expf(sc[i] - mx); den += sc[i]; }
        const float r = 1.0f / den;
        for (int i = 0; i < SL; ++i) sc[i] *= r;
    }
    __syncthreads();
    float sg = 0.0f;
    for (int i = 0; i < SL; ++i) sg += sc[i] * hc[(size_t)i * DIM + t];
    REPCAT[(size_t)b * 256 + t] = sg;
    REPCAT[(size_t)b * 256 + 128 + t] = last[t];
}

// ---------------------------------------------------------------------------
extern "C" void kernel_launch(void* const* d_in, const int* in_sizes, int n_in,
                              void* d_out, int out_size, void* d_ws, size_t ws_size,
                              hipStream_t stream)
{
    (void)in_sizes; (void)n_in; (void)out_size; (void)ws_size;
    const int*   seq      = (const int*)  d_in[0];
    const float* item_emb = (const float*)d_in[1];
    const float* pos_emb  = (const float*)d_in[2];
    const float* ln0_g = (const float*)d_in[3],  *ln0_b = (const float*)d_in[4];
    const float* Wq = (const float*)d_in[5],  *bq = (const float*)d_in[6];
    const float* Wk = (const float*)d_in[7],  *bk = (const float*)d_in[8];
    const float* Wv = (const float*)d_in[9],  *bv = (const float*)d_in[10];
    const float* Wo = (const float*)d_in[11], *bo = (const float*)d_in[12];
    const float* ln1_g = (const float*)d_in[13], *ln1_b = (const float*)d_in[14];
    const float* W1 = (const float*)d_in[15], *b1 = (const float*)d_in[16];
    const float* W2 = (const float*)d_in[17], *b2 = (const float*)d_in[18];
    const float* ln2_g = (const float*)d_in[19], *ln2_b = (const float*)d_in[20];
    const float* gW_in  = (const float*)d_in[21];
    const float* gW_out = (const float*)d_in[22];
    const float* gW_z = (const float*)d_in[23], *gb_z = (const float*)d_in[24];
    const float* gU_z = (const float*)d_in[25];
    const float* gW_r = (const float*)d_in[26], *gb_r = (const float*)d_in[27];
    const float* gU_r = (const float*)d_in[28];
    const float* gW_h = (const float*)d_in[29], *gb_h = (const float*)d_in[30];
    const float* gU_h = (const float*)d_in[31];
    const float* gate_W = (const float*)d_in[32], *gate_b = (const float*)d_in[33];
    const float* out_W  = (const float*)d_in[34], *out_b  = (const float*)d_in[35];
    float* out = (float*)d_out;

    // ---- workspace arena (floats) ----
    float* ws = (float*)d_ws;
    size_t off = 0;
    float* X    = ws + off; off += (size_t)BLD;        // persistent activations
    float* BIG  = ws + off; off += (size_t)4 * BLD;    // reusable B*L x 512 scratch
    float* G    = ws + off; off += (size_t)BLD;        // gout
    float* Hh   = ws + off; off += (size_t)BLD;        // GGNN h / hc
    float* Mb   = ws + off; off += (size_t)BLD;        // temp / GGNN m / ht
    float* Ain  = ws + off; off += (size_t)BL * SL;
    float* Aout = ws + off; off += (size_t)BL * SL;
    float* RC   = ws + off; off += (size_t)BATCH * 256; // [sg|last]
    float* RP   = ws + off; off += (size_t)BATCH * DIM; // reps
    int* uniq = (int*)(ws + off); off += BL;
    int* invb = (int*)(ws + off); off += BL;
    float* s0 = BIG, *s1 = BIG + BLD, *s2 = BIG + 2 * (size_t)BLD, *s3 = BIG + 3 * (size_t)BLD;
    float* CAT = s2;                                    // B*L x 256, overlaps s2..s3

    auto gemm = [&](const float* A, const float* Bm, const float* bias,
                    const float* addC, float* C, int M, int N, int K,
                    int act, int btrans) {
        const int tiles = (M >> 4) * ((N + 15) >> 4);
        const int blocks = (tiles + 7) / 8;
        if (btrans)
            gemm_wmma<128, true ><<<blocks, 256, 0, stream>>>(A, Bm, bias, addC, C, M, N, act);
        else if (K == 128)
            gemm_wmma<128, false><<<blocks, 256, 0, stream>>>(A, Bm, bias, addC, C, M, N, act);
        else if (K == 256)
            gemm_wmma<256, false><<<blocks, 256, 0, stream>>>(A, Bm, bias, addC, C, M, N, act);
        else
            gemm_wmma<512, false><<<blocks, 256, 0, stream>>>(A, Bm, bias, addC, C, M, N, act);
    };
    const int EWB = 256;

    // 1) x = ln0(item_emb[seq] + pos_emb)
    embed_ln0<<<BL, DIM, 0, stream>>>(seq, item_emb, pos_emb, ln0_g, ln0_b, X);

    // 2) q,k,v projections (WMMA)
    gemm(X, Wq, bq, nullptr, s0, BL, DIM, DIM, 0, 0);
    gemm(X, Wk, bk, nullptr, s1, BL, DIM, DIM, 0, 0);
    gemm(X, Wv, bv, nullptr, s2, BL, DIM, DIM, 0, 0);

    // 3) attention -> s3
    attn_kernel<<<BATCH * NHD, 64, 0, stream>>>(s0, s1, s2, s3);

    // 4) mo = attn @ Wo + bo ; x = ln1(x + mo)
    gemm(s3, Wo, bo, nullptr, Mb, BL, DIM, DIM, 0, 0);
    ln_res<<<BL, DIM, 0, stream>>>(X, Mb, ln1_g, ln1_b, X);

    // 5) FFN: hid = relu(x@W1+b1); ff = hid@W2+b2; gout = ln2(x + ff)
    gemm(X, W1, b1, nullptr, BIG, BL, 4 * DIM, DIM, 1, 0);
    gemm(BIG, W2, b2, nullptr, Mb, BL, DIM, 4 * DIM, 0, 0);
    ln_res<<<BL, DIM, 0, stream>>>(X, Mb, ln2_g, ln2_b, G);

    // 6) GGNN graph construction + node embedding gather
    ggnn_build<<<BATCH, 64, 0, stream>>>(seq, uniq, invb, Ain, Aout);
    gather_emb<<<(BLD + EWB - 1) / EWB, EWB, 0, stream>>>(uniq, item_emb, Hh, BLD);

    // 7) m = A_in@(h@gW_in) + A_out@(h@gW_out)
    gemm(Hh, gW_in,  nullptr, nullptr, s0, BL, DIM, DIM, 0, 0);
    gemm(Hh, gW_out, nullptr, nullptr, s1, BL, DIM, DIM, 0, 0);
    propagate<<<BL, DIM, 0, stream>>>(Ain, Aout, s0, s1, Mb);

    // 8) GRU gates (each gate = two WMMA GEMMs with fused add+activation)
    gemm(Mb, gW_z, gb_z, nullptr, s2, BL, DIM, DIM, 0, 0);
    gemm(Hh, gU_z, nullptr, s2, s0, BL, DIM, DIM, 2, 0);       // z  (sigmoid)
    gemm(Mb, gW_r, gb_r, nullptr, s2, BL, DIM, DIM, 0, 0);
    gemm(Hh, gU_r, nullptr, s2, s1, BL, DIM, DIM, 2, 0);       // r  (sigmoid)
    ew_mul<<<(BLD + EWB - 1) / EWB, EWB, 0, stream>>>(s1, Hh, s2, BLD);   // r*h
    gemm(Mb, gW_h, gb_h, nullptr, s3, BL, DIM, DIM, 0, 0);
    gemm(s2, gU_h, nullptr, s3, Mb, BL, DIM, DIM, 3, 0);       // ht (tanh)
    ew_gru<<<(BLD + EWB - 1) / EWB, EWB, 0, stream>>>(s0, Mb, Hh, BLD);   // h update

    // 9) hl = h[inv]; gate = sigmoid([hl|gout]@gate_W + gate_b); hc = lerp
    gather_rows<<<(BLD + EWB - 1) / EWB, EWB, 0, stream>>>(invb, Hh, s1, BLD);
    concat_cat<<<(2 * BLD + EWB - 1) / EWB, EWB, 0, stream>>>(s1, G, CAT, 2 * BLD);
    gemm(CAT, gate_W, gate_b, nullptr, s0, BL, DIM, 2 * DIM, 2, 0);
    ew_lerp<<<(BLD + EWB - 1) / EWB, EWB, 0, stream>>>(s0, s1, G, Hh, BLD); // hc -> Hh

    // 10) session attention readout -> [sg|last]; reps = RC@out_W + out_b
    sess_att<<<BATCH, DIM, 0, stream>>>(Hh, RC);
    gemm(RC, out_W, out_b, nullptr, RP, BATCH, DIM, 2 * DIM, 0, 0);

    // 11) logits = reps @ item_emb^T   (M=512, N=100001, K=128; B transposed)
    gemm(RP, item_emb, nullptr, nullptr, out, BATCH, NV_ITEMS + 1, DIM, 0, 1);
}
